// Attention_2164663517583
// MI455X (gfx1250) — compile-verified, matrix operand order checked by reading, not run on hardware
//
#include <hip/hip_runtime.h>
#include <hip/hip_bf16.h>
#include <math.h>

#define HIDDEN 1024
#define BATCH 32
#define MAXLEN 2048
#define NEG_INF (-1e10f)

#define LOG2E    1.4426950408889634f
#define TWOLOG2E 2.8853900817779268f

typedef __attribute__((ext_vector_type(16))) __bf16 v16bf;
typedef __attribute__((ext_vector_type(8)))  float  v8f;

union FragBF { uint4 q[2]; v16bf v; };

__device__ __forceinline__ unsigned short f2bf(float x) {
    unsigned u = __float_as_uint(x);
    u = (u + 0x7FFFu + ((u >> 16) & 1u)) >> 16;   // round-to-nearest-even
    return (unsigned short)u;
}

// Branchless tanh on the HW TRANS pipe: tanh(x) = 1 - 2/(exp2(2*log2e*x)+1).
// v_exp_f32 + v_rcp_f32; saturates to +-1 for large |x|; co-executes with XDL.
__device__ __forceinline__ float fast_tanh(float x) {
    float e = __builtin_amdgcn_exp2f(x * TWOLOG2E);
    float r = __builtin_amdgcn_rcpf(e + 1.0f);
    return fmaf(-2.0f, r, 1.0f);
}

// ---------------------------------------------------------------------------
// Kernel 0: Ua_w f32 -> bf16, pre-swizzled into WMMA B-fragment layout.
// Element (n,k): hi=(k/8)%2, blk=k/16, w=k%8 -> lane=hi*16+n, half=blk*8+w
// (mirrors ISA 7.12.2 16-bit fragment table). One 1KB block per (chunk, t).
// ---------------------------------------------------------------------------
__global__ void k_convert_B(const float* __restrict__ Ua_w,
                            unsigned char* __restrict__ wsB) {
    int idx = blockIdx.x * 256 + threadIdx.x;   // 1M elems / 4
    int q = idx & 255;
    int o = idx >> 8;
    int k0 = q * 4;
    const float4 f = *(const float4*)(Ua_w + (size_t)o * HIDDEN + k0);
    int c  = o >> 4,  n  = o & 15;
    int t  = k0 >> 5, kk = k0 & 31;
    int hi = (kk >> 3) & 1, blk = kk >> 4, w = kk & 7;
    int lane = hi * 16 + n;
    int hs   = blk * 8 + w;
    uint2 d;
    d.x = (unsigned)f2bf(f.x) | ((unsigned)f2bf(f.y) << 16);
    d.y = (unsigned)f2bf(f.z) | ((unsigned)f2bf(f.w) << 16);
    *(uint2*)(wsB + (((size_t)(c * 32 + t)) << 10) + lane * 32 + hs * 2) = d;
}

// ---------------------------------------------------------------------------
// Kernel 1: qva[b][o] = { query[b]·Wa_w[o] + Wa_b[o] + Ua_b[o],  va_w[o] }
// Packed so the score-kernel epilogue does one b64 load per chunk.
// ---------------------------------------------------------------------------
__global__ void k_qsum(const float* __restrict__ query,
                       const float* __restrict__ Wa_w,
                       const float* __restrict__ Wa_b,
                       const float* __restrict__ Ua_b,
                       const float* __restrict__ va_w,
                       float2* __restrict__ qva) {
    int o = blockIdx.x * 256 + threadIdx.x;
    int b = blockIdx.y;
    const float4* qr = (const float4*)(query + (size_t)b * HIDDEN);
    const float4* wr = (const float4*)(Wa_w + (size_t)o * HIDDEN);
    float4 a = {0.f, 0.f, 0.f, 0.f};
    for (int i = 0; i < HIDDEN / 4; ++i) {
        float4 x = qr[i], y = wr[i];
        a.x += x.x * y.x; a.y += x.y * y.y; a.z += x.z * y.z; a.w += x.w * y.w;
    }
    float2 r;
    r.x = (a.x + a.y) + (a.z + a.w) + Wa_b[o] + Ua_b[o];
    r.y = va_w[o];
    qva[(size_t)b * HIDDEN + o] = r;
}

// ---------------------------------------------------------------------------
// Kernel 2: fused  score[b][l] = tanh(qsum[b] + key[b,l]·Ua^T) · va_w  (+mask)
// WG = (b, 128 l-rows). 8 waves, 256KB dynamic LDS (key tile, bf16,
// A-fragment layout). Per wave: 16-row M tile, o-chunks in pairs for two
// independent WMMA accumulator chains.
// ---------------------------------------------------------------------------
__global__ void __launch_bounds__(256, 1)
k_score(const float* __restrict__ key,
        const unsigned char* __restrict__ wsB,
        const float2* __restrict__ qva,
        const float* __restrict__ va_b,
        const int* __restrict__ maskp,
        float* __restrict__ score) {
    extern __shared__ unsigned char smem[];     // 8 tiles * 32KB = 256KB
    const int tid = threadIdx.x;
    const int b   = blockIdx.y;
    const int l0  = blockIdx.x * 128;

    // ---- stage key tile (128 x 1024 f32) -> LDS bf16, A-fragment swizzle
    const float* keyB = key + ((size_t)b * MAXLEN + l0) * HIDDEN;
    for (int i = 0; i < 128; ++i) {
        int idx = tid + i * 256;
        int q = idx & 255, r = idx >> 8;
        int k0 = q * 4;
        float4 f = *(const float4*)(keyB + (size_t)r * HIDDEN + k0);
        int tile = r >> 4, rl = r & 15;
        int t  = k0 >> 5, kk = k0 & 31;
        int hi = (kk >> 3) & 1, blk = kk >> 4, w = kk & 7;
        int lane = hi * 16 + rl;
        int hs   = blk * 8 + w;
        uint2 d;
        d.x = (unsigned)f2bf(f.x) | ((unsigned)f2bf(f.y) << 16);
        d.y = (unsigned)f2bf(f.z) | ((unsigned)f2bf(f.w) << 16);
        *(uint2*)(smem + tile * 32768 + t * 1024 + lane * 32 + hs * 2) = d;
    }
    __syncthreads();

    const int wave = tid >> 5, lane = tid & 31;
    const int nloc = lane & 15, half = lane >> 4;
    const unsigned char* ldsA = smem + wave * 32768;
    const float2* qv = qva + (size_t)b * HIDDEN;

    float sp[8];
#pragma unroll
    for (int j = 0; j < 8; ++j) sp[j] = 0.f;

    for (int c = 0; c < 64; c += 2) {
        v8f acc0 = {0.f,0.f,0.f,0.f,0.f,0.f,0.f,0.f};
        v8f acc1 = {0.f,0.f,0.f,0.f,0.f,0.f,0.f,0.f};
        const unsigned char* bp0 = wsB + (((size_t)c * 32) << 10) + lane * 32;
        const unsigned char* bp1 = bp0 + (32u << 10);
        __builtin_prefetch((const void*)(bp1 + (32u << 10)), 0, 1);  // next pair
#pragma unroll 4
        for (int t = 0; t < 32; ++t) {
            FragBF a, f0, f1;
            const uint4* ap = (const uint4*)(ldsA + t * 1024 + lane * 32);
            a.q[0] = ap[0]; a.q[1] = ap[1];
            const uint4* p0 = (const uint4*)(bp0 + t * 1024);
            const uint4* p1 = (const uint4*)(bp1 + t * 1024);
            f0.q[0] = p0[0]; f0.q[1] = p0[1];
            f1.q[0] = p1[0]; f1.q[1] = p1[1];
            acc0 = __builtin_amdgcn_wmma_f32_16x16x32_bf16(
                false, a.v, false, f0.v, (short)0, acc0, false, false);
            acc1 = __builtin_amdgcn_wmma_f32_16x16x32_bf16(
                false, a.v, false, f1.v, (short)0, acc1, false, false);
        }
        // epilogue: branchless tanh on TRANS pipe, fold into score partials
        float2 q0 = qv[c * 16 + nloc];
        float2 q1 = qv[c * 16 + 16 + nloc];
#pragma unroll
        for (int j = 0; j < 8; ++j) sp[j] = fmaf(fast_tanh(acc0[j] + q0.x), q0.y, sp[j]);
#pragma unroll
        for (int j = 0; j < 8; ++j) sp[j] = fmaf(fast_tanh(acc1[j] + q1.x), q1.y, sp[j]);
    }

    // reduce over the 16 N-lanes within each half-wave
#pragma unroll
    for (int j = 0; j < 8; ++j) {
        float v = sp[j];
        v += __shfl_xor(v, 1);
        v += __shfl_xor(v, 2);
        v += __shfl_xor(v, 4);
        v += __shfl_xor(v, 8);
        sp[j] = v;
    }
    if (nloc == 0) {
        float vab = va_b[0];
#pragma unroll
        for (int j = 0; j < 8; ++j) {
            int l = l0 + wave * 16 + half * 8 + j;   // lanes16-31 hold M=j+8
            int mk = maskp[(size_t)b * MAXLEN + l];
            score[(size_t)b * MAXLEN + l] = mk ? (sp[j] + vab) : NEG_INF;
        }
    }
}

// ---------------------------------------------------------------------------
// Kernel 3: softmax over L + ctx[b,h] = sum_l attn[b,l]*value[b,l,h]
// ---------------------------------------------------------------------------
__global__ void k_softmax_ctx(const float* __restrict__ score,
                              const float* __restrict__ value,
                              float* __restrict__ out) {
    __shared__ float p[MAXLEN];
    __shared__ float red[256];
    int b = blockIdx.x, tid = threadIdx.x;
    const float* s = score + (size_t)b * MAXLEN;

    float mx = -3.4e38f;
    for (int i = 0; i < 8; ++i) mx = fmaxf(mx, s[tid + i * 256]);
    red[tid] = mx; __syncthreads();
    for (int st = 128; st > 0; st >>= 1) {
        if (tid < st) red[tid] = fmaxf(red[tid], red[tid + st]);
        __syncthreads();
    }
    mx = red[0]; __syncthreads();

    float sum = 0.f;
    for (int i = 0; i < 8; ++i) {
        int l = tid + i * 256;
        float e = __builtin_amdgcn_exp2f((s[l] - mx) * LOG2E);  // arg <= 0: no range issues
        p[l] = e; sum += e;
    }
    red[tid] = sum; __syncthreads();
    for (int st = 128; st > 0; st >>= 1) {
        if (tid < st) red[tid] += red[tid + st];
        __syncthreads();
    }
    float inv = 1.0f / red[0];

    const float4* v = (const float4*)(value + (size_t)b * MAXLEN * HIDDEN);
    float4 acc = {0.f, 0.f, 0.f, 0.f};
    for (int l = 0; l < MAXLEN; ++l) {
        float w = p[l] * inv;
        float4 x = v[(size_t)l * 256 + tid];
        acc.x = fmaf(w, x.x, acc.x); acc.y = fmaf(w, x.y, acc.y);
        acc.z = fmaf(w, x.z, acc.z); acc.w = fmaf(w, x.w, acc.w);
    }
    *(float4*)(out + (size_t)b * HIDDEN + tid * 4) = acc;
}

// ---------------------------------------------------------------------------
extern "C" void kernel_launch(void* const* d_in, const int* in_sizes, int n_in,
                              void* d_out, int out_size, void* d_ws, size_t ws_size,
                              hipStream_t stream) {
    const float* query = (const float*)d_in[0];
    const float* key   = (const float*)d_in[1];
    const float* value = (const float*)d_in[2];
    const float* Wa_w  = (const float*)d_in[3];
    const float* Wa_b  = (const float*)d_in[4];
    const float* Ua_w  = (const float*)d_in[5];
    const float* Ua_b  = (const float*)d_in[6];
    const float* va_w  = (const float*)d_in[7];
    const float* va_b  = (const float*)d_in[8];
    const int*   mask  = (const int*)d_in[9];

    unsigned char* wsB = (unsigned char*)d_ws;                                    // 2 MB bf16 Ua_w
    float2* qva  = (float2*)((unsigned char*)d_ws + (2u << 20));                  // 256 KB
    float*  scor = (float*)((unsigned char*)d_ws + (2u << 20) + (256u << 10));    // 256 KB
    float*  out  = (float*)d_out;

    k_convert_B  <<<dim3(1024),   dim3(256), 0,      stream>>>(Ua_w, wsB);
    k_qsum       <<<dim3(4, 32),  dim3(256), 0,      stream>>>(query, Wa_w, Wa_b, Ua_b, va_w, qva);
    k_score      <<<dim3(16, 32), dim3(256), 262144, stream>>>(key, wsB, qva, va_b, mask, scor);
    k_softmax_ctx<<<dim3(32),     dim3(256), 0,      stream>>>(scor, value, out);
}
